// PointTrans_Layer_67920612819552
// MI455X (gfx1250) — matrix operand order repaired and checked by reading, not compiled, not run on hardware
//
#include <hip/hip_runtime.h>
#include <hip/hip_bf16.h>

typedef __attribute__((ext_vector_type(16))) __bf16 v16bf;
typedef __attribute__((ext_vector_type(8)))  __bf16 bf8;
typedef __attribute__((ext_vector_type(8)))  float  v8f;

// ---------------------------------------------------------------------------
// helpers
// ---------------------------------------------------------------------------
__device__ __forceinline__ v8f wmma_bf16(v16bf a, v16bf b, v8f c) {
  return __builtin_amdgcn_wmma_f32_16x16x32_bf16(false, a, false, b, (short)0, c,
                                                 false, false);
}

// A-fragment for 16x16x32 bf16 WMMA. Row-major source [16 x stride] bf16.
// lane l: row m = l&15; K-run of 16 at k0 (k0 = kc*32 + ((l&16)?16:0)).
__device__ __forceinline__ v16bf load_afrag(const __bf16* row, int k0) {
  bf8 lo = *(const bf8*)(row + k0);
  bf8 hi = *(const bf8*)(row + k0 + 8);
  v16bf a;
#pragma unroll
  for (int i = 0; i < 8; ++i) { a[i] = lo[i]; a[i + 8] = hi[i]; }
  return a;
}

// order-preserving float->uint encoding for atomicMax
__device__ __forceinline__ unsigned enc_f32(float f) {
  unsigned u = __float_as_uint(f);
  return (u & 0x80000000u) ? ~u : (u | 0x80000000u);
}
__device__ __forceinline__ float dec_f32(unsigned u) {
  u = (u & 0x80000000u) ? (u & 0x7fffffffu) : ~u;
  return __uint_as_float(u);
}

// CDNA5 async global->LDS copy (16B per lane), tracked by ASYNCcnt.
__device__ __forceinline__ void async_copy16(const void* gptr, void* lptr) {
  unsigned lds = (unsigned)(uintptr_t)lptr;          // LDS byte offset
  unsigned long long ga = (unsigned long long)(uintptr_t)gptr;
  asm volatile("global_load_async_to_lds_b128 %0, %1, off"
               :: "v"(lds), "v"(ga) : "memory");
}
__device__ __forceinline__ void async_wait0() {
  asm volatile("s_wait_asynccnt 0x0" ::: "memory");
}

// ---------------------------------------------------------------------------
// weight prep: build bf16 transposed weights in ws
// layout (each 2048 bf16): [0]=w_linT [1]=w_srcT [2]=w_dstT [3]=pos_w2T
//                          [4]=att_w1T [5]=att_w2T
// For [K,Nout] weights, T is [Nout][K] row-major.
// ---------------------------------------------------------------------------
__global__ void prep_weights(const float* __restrict__ wlin,
                             const float* __restrict__ wsrc,
                             const float* __restrict__ wdst,
                             const float* __restrict__ posw2,
                             const float* __restrict__ attw1,
                             const float* __restrict__ attw2,
                             __bf16* __restrict__ wT) {
  for (int i = threadIdx.x; i < 2048; i += blockDim.x) {
    int nn = i >> 6, kk = i & 63;               // [32][64] targets, src [64][32]
    wT[0 * 2048 + i] = (__bf16)wlin[kk * 32 + nn];
    wT[1 * 2048 + i] = (__bf16)wsrc[kk * 32 + nn];
    wT[2 * 2048 + i] = (__bf16)wdst[kk * 32 + nn];
    wT[3 * 2048 + i] = (__bf16)posw2[kk * 32 + nn];
    wT[5 * 2048 + i] = (__bf16)attw2[kk * 32 + nn];
    int n2 = i >> 5, k2 = i & 31;               // att_w1: [32][64] -> T [64][32]
    wT[4 * 2048 + i] = (__bf16)attw1[k2 * 64 + n2];
  }
}

// ---------------------------------------------------------------------------
__global__ void init_buffers(float* __restrict__ denom, unsigned* __restrict__ amax,
                             float* __restrict__ out, float* __restrict__ bn, int n32) {
  int i = blockIdx.x * blockDim.x + threadIdx.x;
  if (i < n32) { denom[i] = 0.f; amax[i] = 0u; out[i] = 0.f; }
  if (i < 512) bn[i] = 0.f;
}

__global__ void build_edges(const int* __restrict__ ei, int E, int EP,
                            int* __restrict__ src32, int* __restrict__ dst32) {
  int i = blockIdx.x * blockDim.x + threadIdx.x;
  if (i >= EP) return;
  if (i < E) { src32[i] = ei[i]; dst32[i] = ei[E + i]; }
  else       { src32[i] = i - E; dst32[i] = i - E; }      // self loops
}

__global__ void bn_finalize(const float* __restrict__ sum, const float* __restrict__ sq,
                            const float* __restrict__ g, const float* __restrict__ beta,
                            float* __restrict__ ss, float invM) {
  int i = threadIdx.x;  // 64
  float mu  = sum[i] * invM;
  float var = sq[i] * invM - mu * mu;
  float s   = g[i] * rsqrtf(var + 1e-5f);
  ss[i]      = s;
  ss[64 + i] = beta[i] - mu * s;
}

// ---------------------------------------------------------------------------
// node GEMMs: x[N,64] @ {w_lin,w_src,w_dst}[64,32]  (bf16 WMMA, f32 accum)
// wave per 16 nodes, 4 waves / block
// ---------------------------------------------------------------------------
__global__ void node_gemm(const float* __restrict__ x, const __bf16* __restrict__ wT,
                          float* __restrict__ xval, float* __restrict__ asrc,
                          float* __restrict__ adst, int n) {
  const int lane = threadIdx.x & 31, wv = threadIdx.x >> 5;
  const int l16 = lane & 15, hi = (lane >> 4) & 1;
  __shared__ alignas(32) __bf16 sw[3][2048];
  for (int i = threadIdx.x * 16; i < 4096; i += 128 * 16) {
    async_copy16((const char*)wT + i,        (char*)&sw[0][0] + i);
    async_copy16((const char*)wT + 4096 + i, (char*)&sw[1][0] + i);
    async_copy16((const char*)wT + 8192 + i, (char*)&sw[2][0] + i);
  }
  async_wait0();
  __syncthreads();

  const int base = (blockIdx.x * 4 + wv) * 16;
  int m = base + l16; if (m >= n) m = n - 1;
  const float* xr = x + (size_t)m * 64;

  v16bf A[2];
#pragma unroll
  for (int kc = 0; kc < 2; ++kc) {
    int k0 = kc * 32 + hi * 16;
    float4 f0 = *(const float4*)(xr + k0);
    float4 f1 = *(const float4*)(xr + k0 + 4);
    float4 f2 = *(const float4*)(xr + k0 + 8);
    float4 f3 = *(const float4*)(xr + k0 + 12);
    v16bf a;
    a[0]  = (__bf16)f0.x; a[1]  = (__bf16)f0.y; a[2]  = (__bf16)f0.z; a[3]  = (__bf16)f0.w;
    a[4]  = (__bf16)f1.x; a[5]  = (__bf16)f1.y; a[6]  = (__bf16)f1.z; a[7]  = (__bf16)f1.w;
    a[8]  = (__bf16)f2.x; a[9]  = (__bf16)f2.y; a[10] = (__bf16)f2.z; a[11] = (__bf16)f2.w;
    a[12] = (__bf16)f3.x; a[13] = (__bf16)f3.y; a[14] = (__bf16)f3.z; a[15] = (__bf16)f3.w;
    A[kc] = a;
  }

  const bool full = (base + 16) <= n;            // wave-uniform fast path
  float* outs[3] = {xval, asrc, adst};
#pragma unroll
  for (int w = 0; w < 3; ++w) {
#pragma unroll
    for (int t = 0; t < 2; ++t) {
      int ch = t * 16 + l16;
      v8f c = {};
#pragma unroll
      for (int kc = 0; kc < 2; ++kc) {
        v16bf bw = *(const v16bf*)&sw[w][ch * 64 + kc * 32 + hi * 16];
        c = wmma_bf16(A[kc], bw, c);
      }
      float* op = outs[w] + (size_t)(base + hi * 8) * 32 + ch;
      if (full) {
#pragma unroll
        for (int r = 0; r < 8; ++r) op[(size_t)r * 32] = c[r];
      } else {
#pragma unroll
        for (int r = 0; r < 8; ++r)
          if (base + hi * 8 + r < n) op[(size_t)r * 32] = c[r];
      }
    }
  }
}

// ---------------------------------------------------------------------------
// edge passes. PASS: 1 = pos-BN stats, 2 = att-BN stats, 3 = segment max,
//                    4 = sum of exp,   5 = final weighted scatter
// wave per 16 edges, 4 waves / block; full MLP recompute each pass (all node
// data is L2-resident on MI455X's 192MB L2, so recompute beats materialize).
// ---------------------------------------------------------------------------
template <int PASS>
__global__ void edge_pass(const int* __restrict__ src32, const int* __restrict__ dst32,
                          int EP,
                          const float* __restrict__ pos,
                          const float* __restrict__ a_src, const float* __restrict__ a_dst,
                          const float* __restrict__ x_val,
                          const float* __restrict__ posw1, const float* __restrict__ posb1,
                          const __bf16* __restrict__ wT,
                          const float* __restrict__ posb2, const float* __restrict__ attb1,
                          const float* __restrict__ attb2,
                          const float* __restrict__ pos_ss, const float* __restrict__ att_ss,
                          float* __restrict__ bn_out,
                          unsigned* __restrict__ amax, float* __restrict__ denom,
                          float* __restrict__ out) {
  const int tid = threadIdx.x;
  const int lane = tid & 31, wv = tid >> 5;
  const int l16 = lane & 15, hi = (lane >> 4) & 1;
  const int tile = blockIdx.x * 4 + wv;
  const bool tile_full = (tile * 16 + 16) <= EP;   // wave-uniform

  __shared__ alignas(32) __bf16 sh_w2pos[32 * 64];
  __shared__ alignas(32) __bf16 sh_w1att[64 * 32];
  __shared__ alignas(32) __bf16 sh_w2att[32 * 64];
  __shared__ float sh_w1pos[256];          // [3][64] + b1
  __shared__ float sh_posbn[128], sh_attbn[128];
  __shared__ float sh_b2pos[32], sh_b1att[64], sh_b2att[32];
  __shared__ float sh_bnacc[128];
  __shared__ alignas(32) __bf16 sh_h[4][16 * 64];
  __shared__ alignas(16) float  sh_delta[4][16 * 32];
  __shared__ alignas(32) __bf16 sh_attin[4][16 * 32];
  __shared__ float sh_diff[4][16 * 4];
  __shared__ int sh_src[4][16], sh_dst[4][16];

  // async global->LDS staging of bf16 weights (ASYNCcnt path)
  for (int i = tid * 16; i < 4096; i += 128 * 16) {
    async_copy16((const char*)(wT + 3 * 2048) + i, (char*)&sh_w2pos[0] + i);
    async_copy16((const char*)(wT + 4 * 2048) + i, (char*)&sh_w1att[0] + i);
    async_copy16((const char*)(wT + 5 * 2048) + i, (char*)&sh_w2att[0] + i);
  }
  for (int i = tid; i < 256; i += 128) sh_w1pos[i] = (i < 192) ? posw1[i] : posb1[i - 192];
  if (tid < 128) {
    sh_bnacc[tid] = 0.f;
    if constexpr (PASS >= 2) sh_posbn[tid] = pos_ss[tid];
    if constexpr (PASS >= 3) sh_attbn[tid] = att_ss[tid];
  }
  if (tid < 32) { sh_b2pos[tid] = posb2[tid]; sh_b2att[tid] = attb2[tid]; }
  if (tid < 64) sh_b1att[tid] = attb1[tid];

  if (lane < 16) {
    int e = tile * 16 + lane;
    int s = -1, d = 0;
    if (e < EP) { s = src32[e]; d = dst32[e]; }
    sh_src[wv][lane] = s;
    sh_dst[wv][lane] = d;
  }
  async_wait0();
  __syncthreads();

  // ---- gather pos diffs (L2-resident) ----
  if (lane < 16) {
    int s = sh_src[wv][lane], d = sh_dst[wv][lane];
    int sc = s < 0 ? 0 : s;
    const float* ps = pos + (size_t)sc * 3;
    const float* pd = pos + (size_t)d * 3;
    sh_diff[wv][lane * 4 + 0] = pd[0] - ps[0];
    sh_diff[wv][lane * 4 + 1] = pd[1] - ps[1];
    sh_diff[wv][lane * 4 + 2] = pd[2] - ps[2];
  }
  __syncthreads();

  // ---- pos MLP layer 1 (K=3, VALU) in C/D layout ----
#pragma unroll
  for (int t = 0; t < 4; ++t) {
    int ch = t * 16 + l16;
    float w0 = sh_w1pos[ch], w1 = sh_w1pos[64 + ch], w2 = sh_w1pos[128 + ch];
    float b = sh_w1pos[192 + ch];
    float ssum = 0.f, ssq = 0.f;
#pragma unroll
    for (int r = 0; r < 8; ++r) {
      int m = r + hi * 8;
      float d0 = sh_diff[wv][m * 4], d1 = sh_diff[wv][m * 4 + 1], d2 = sh_diff[wv][m * 4 + 2];
      float h = fmaf(d0, w0, fmaf(d1, w1, fmaf(d2, w2, b)));
      if constexpr (PASS == 1) {
        if (tile_full || sh_src[wv][m] >= 0) { ssum += h; ssq += h * h; }
      } else {
        h = fmaf(h, sh_posbn[ch], sh_posbn[64 + ch]);
        h = h > 0.f ? h : 0.f;
        sh_h[wv][m * 64 + ch] = (__bf16)h;
      }
    }
    if constexpr (PASS == 1) {
      atomicAdd(&sh_bnacc[ch], ssum);
      atomicAdd(&sh_bnacc[64 + ch], ssq);
    }
  }
  if constexpr (PASS == 1) {
    __syncthreads();
    if (tid < 128) atomicAdd(&bn_out[tid], sh_bnacc[tid]);
    return;
  }
  __syncthreads();

  // ---- pos MLP layer 2: [16x64] @ [64x32] (WMMA bf16) -> delta ----
  {
    v16bf A0 = load_afrag(&sh_h[wv][l16 * 64], 0 * 32 + hi * 16);
    v16bf A1 = load_afrag(&sh_h[wv][l16 * 64], 1 * 32 + hi * 16);
#pragma unroll
    for (int t = 0; t < 2; ++t) {
      int ch = t * 16 + l16;
      float b = sh_b2pos[ch];
      v8f c;
#pragma unroll
      for (int i = 0; i < 8; ++i) c[i] = b;
      c = wmma_bf16(A0, *(const v16bf*)&sh_w2pos[ch * 64 + 0 * 32 + hi * 16], c);
      c = wmma_bf16(A1, *(const v16bf*)&sh_w2pos[ch * 64 + 1 * 32 + hi * 16], c);
#pragma unroll
      for (int r = 0; r < 8; ++r) sh_delta[wv][(r + hi * 8) * 32 + ch] = c[r];
    }
  }
  __syncthreads();

  // ---- att_in = alpha_dst[dst] - alpha_src[src] + delta  (bf16) ----
  {
    int m = l16;
    int s = sh_src[wv][m], d = sh_dst[wv][m];
    int sc = s < 0 ? 0 : s;
    const float* pa = a_dst + (size_t)d * 32 + hi * 16;
    const float* pb = a_src + (size_t)sc * 32 + hi * 16;
    const float* pdl = &sh_delta[wv][m * 32 + hi * 16];
    __bf16* po = &sh_attin[wv][m * 32 + hi * 16];
#pragma unroll
    for (int c = 0; c < 16; ++c) po[c] = (__bf16)(pa[c] - pb[c] + pdl[c]);
  }
  __syncthreads();

  // ---- att MLP layer 1: [16x32] @ [32x64] (WMMA bf16) ----
  v8f t1[4];
  {
    v16bf A = load_afrag(&sh_attin[wv][l16 * 32], hi * 16);
#pragma unroll
    for (int t = 0; t < 4; ++t) {
      int ch = t * 16 + l16;
      float b = sh_b1att[ch];
      v8f c;
#pragma unroll
      for (int i = 0; i < 8; ++i) c[i] = b;
      t1[t] = wmma_bf16(A, *(const v16bf*)&sh_w1att[ch * 32 + hi * 16], c);
    }
  }

  if constexpr (PASS == 2) {
#pragma unroll
    for (int t = 0; t < 4; ++t) {
      int ch = t * 16 + l16;
      float ssum = 0.f, ssq = 0.f;
#pragma unroll
      for (int r = 0; r < 8; ++r) {
        int m = r + hi * 8;
        if (tile_full || sh_src[wv][m] >= 0) { float h = t1[t][r]; ssum += h; ssq += h * h; }
      }
      atomicAdd(&sh_bnacc[ch], ssum);
      atomicAdd(&sh_bnacc[64 + ch], ssq);
    }
    __syncthreads();
    if (tid < 128) atomicAdd(&bn_out[tid], sh_bnacc[tid]);
    return;
  } else {
    // ---- att BN + ReLU -> bf16 staging (reuse sh_h) ----
    __syncthreads();
#pragma unroll
    for (int t = 0; t < 4; ++t) {
      int ch = t * 16 + l16;
#pragma unroll
      for (int r = 0; r < 8; ++r) {
        float h = fmaf(t1[t][r], sh_attbn[ch], sh_attbn[64 + ch]);
        h = h > 0.f ? h : 0.f;
        sh_h[wv][(r + hi * 8) * 64 + ch] = (__bf16)h;
      }
    }
    __syncthreads();

    // ---- att MLP layer 2: [16x64] @ [64x32] (WMMA bf16) -> a_pre ----
    v16bf A0 = load_afrag(&sh_h[wv][l16 * 64], 0 * 32 + hi * 16);
    v16bf A1 = load_afrag(&sh_h[wv][l16 * 64], 1 * 32 + hi * 16);
#pragma unroll
    for (int t = 0; t < 2; ++t) {
      int ch = t * 16 + l16;
      float b = sh_b2att[ch];
      v8f c;
#pragma unroll
      for (int i = 0; i < 8; ++i) c[i] = b;
      c = wmma_bf16(A0, *(const v16bf*)&sh_w2att[ch * 64 + 0 * 32 + hi * 16], c);
      c = wmma_bf16(A1, *(const v16bf*)&sh_w2att[ch * 64 + 1 * 32 + hi * 16], c);

      auto body = [&](int r) {
        int m = r + hi * 8;
        int s = sh_src[wv][m];
        int d = sh_dst[wv][m];
        size_t o = (size_t)d * 32 + ch;
        float apre = c[r];
        if constexpr (PASS == 3) {
          atomicMax(&amax[o], enc_f32(apre));
        } else {
          float am = dec_f32(amax[o]);
          float e = __expf(apre - am);
          if constexpr (PASS == 4) {
            atomicAdd(&denom[o], e);
          } else {
            float dn = denom[o] + 1e-16f;
            float xv = x_val[(size_t)s * 32 + ch];
            float msg = (e / dn) * (xv + sh_delta[wv][m * 32 + ch]);
            atomicAdd(&out[o], msg);
          }
        }
      };
      if (tile_full) {               // wave-uniform: no per-row predication
#pragma unroll
        for (int r = 0; r < 8; ++r) body(r);
      } else {
#pragma unroll
        for (int r = 0; r < 8; ++r)
          if (sh_src[wv][r + hi * 8] >= 0) body(r);
      }
    }
  }
}

// ---------------------------------------------------------------------------
extern "C" void kernel_launch(void* const* d_in, const int* in_sizes, int n_in,
                              void* d_out, int out_size, void* d_ws, size_t ws_size,
                              hipStream_t stream) {
  const float* x      = (const float*)d_in[0];
  const float* pos    = (const float*)d_in[1];
  const int*   ei     = (const int*)d_in[2];
  const float* w_lin  = (const float*)d_in[3];
  const float* w_src  = (const float*)d_in[4];
  const float* w_dst  = (const float*)d_in[5];
  const float* posw1  = (const float*)d_in[6];
  const float* posb1  = (const float*)d_in[7];
  const float* posg   = (const float*)d_in[8];
  const float* posbe  = (const float*)d_in[9];
  const float* posw2  = (const float*)d_in[10];
  const float* posb2  = (const float*)d_in[11];
  const float* attw1  = (const float*)d_in[12];
  const float* attb1  = (const float*)d_in[13];
  const float* attg   = (const float*)d_in[14];
  const float* attbe  = (const float*)d_in[15];
  const float* attw2  = (const float*)d_in[16];
  const float* attb2  = (const float*)d_in[17];
  float* out = (float*)d_out;

  const int N  = in_sizes[0] / 64;
  const int E  = in_sizes[2] / 2;
  const int EP = E + N;                // with self-loops
  const int n32 = N * 32;

  // ---- workspace carve-up ----
  char* w = (char*)d_ws;
  auto carve = [&](size_t bytes) { char* p = w; w += (bytes + 255) & ~(size_t)255; return p; };
  float*    x_val  = (float*)carve((size_t)n32 * 4);
  float*    a_src  = (float*)carve((size_t)n32 * 4);
  float*    a_dst  = (float*)carve((size_t)n32 * 4);
  float*    denom  = (float*)carve((size_t)n32 * 4);
  unsigned* amax   = (unsigned*)carve((size_t)n32 * 4);
  float*    bnstat = (float*)carve(512 * 4);       // sums/sq + scale/shift
  int*      src32  = (int*)carve((size_t)EP * 4);
  int*      dst32  = (int*)carve((size_t)EP * 4);
  __bf16*   wT     = (__bf16*)carve(6 * 2048 * 2); // transposed bf16 weights

  float* pos_ss = bnstat + 256;
  float* att_ss = bnstat + 384;

  prep_weights<<<1, 256, 0, stream>>>(w_lin, w_src, w_dst, posw2, attw1, attw2, wT);
  init_buffers<<<(n32 + 255) / 256, 256, 0, stream>>>(denom, amax, out, bnstat, n32);
  build_edges<<<(EP + 255) / 256, 256, 0, stream>>>(ei, E, EP, src32, dst32);
  node_gemm<<<(N + 63) / 64, 128, 0, stream>>>(x, wT, x_val, a_src, a_dst, N);

  const int ntiles = (EP + 15) / 16;
  const int eblk   = (ntiles + 3) / 4;
  const float invM = 1.0f / (float)EP;

#define EDGE_ARGS(BN) src32, dst32, EP, pos, a_src, a_dst, x_val, posw1, posb1, wT, \
                      posb2, attb1, attb2, pos_ss, att_ss, (BN), amax, denom, out
  edge_pass<1><<<eblk, 128, 0, stream>>>(EDGE_ARGS(bnstat));
  bn_finalize<<<1, 64, 0, stream>>>(bnstat, bnstat + 64, posg, posbe, pos_ss, invM);
  edge_pass<2><<<eblk, 128, 0, stream>>>(EDGE_ARGS(bnstat + 128));
  bn_finalize<<<1, 64, 0, stream>>>(bnstat + 128, bnstat + 192, attg, attbe, att_ss, invM);
  edge_pass<3><<<eblk, 128, 0, stream>>>(EDGE_ARGS(bnstat));
  edge_pass<4><<<eblk, 128, 0, stream>>>(EDGE_ARGS(bnstat));
  edge_pass<5><<<eblk, 128, 0, stream>>>(EDGE_ARGS(bnstat));
#undef EDGE_ARGS
}
